// BidirectionAttention_29102698397854
// MI455X (gfx1250) — compile-verified, matrix operand order checked by reading, not run on hardware
//
#include <hip/hip_runtime.h>
#include <hip/hip_bf16.h>

// ---------------------------------------------------------------------------
// BiDAF bidirectional attention, MI455X (gfx1250), wave32 + WMMA bf16.
// B=8, JX=2048, JQ=256, D=512, fp32 I/O, bf16 WMMA with f32 accumulate.
// v2: u pre-converted once to bf16 in BOTH row-major (phase A) and
//     transposed (phase C) layouts -> removes per-wave cvt storms and the
//     16-way strided dword loads; adds global_prefetch on the h stream.
// ---------------------------------------------------------------------------

constexpr int Bc  = 8;
constexpr int JXc = 2048;
constexpr int JQc = 256;
constexpr int Dc  = 512;
constexpr int WAVES = 4;                 // waves per block in kernel 1
constexpr int ROWS_PER_BLOCK = WAVES * 16;

typedef __attribute__((ext_vector_type(16))) __bf16 v16bf;
typedef __attribute__((ext_vector_type(8)))  __bf16 v8bf;
typedef __attribute__((ext_vector_type(8)))  float  v8f;

// ---------------------------------------------------------------------------
// Kernel P: convert u -> bf16 in two layouts:
//   u_bf [b][j][d]  (row-major, phase-A B fragments)
//   ut_bf[b][d][j]  (transposed, phase-C B fragments)
// 4.2 MB read, 4 MB write: noise vs the 175 MB roofline; both stay in L2.
// ---------------------------------------------------------------------------
__global__ __launch_bounds__(256) void convert_u_kernel(
    const float* __restrict__ u, __bf16* __restrict__ u_bf,
    __bf16* __restrict__ ut_bf)
{
    const int e = blockIdx.x * 256 + threadIdx.x;        // B*JQ*D = 1,048,576
    const int b = e / (JQc * Dc);
    const int r = e % (JQc * Dc);
    const int j = r / Dc;
    const int d = r % Dc;
    const __bf16 bv = (__bf16)u[e];
    u_bf[e] = bv;                                        // coalesced
    ut_bf[((size_t)b * Dc + d) * JQc + j] = bv;          // strided (tiny total)
}

// ---------------------------------------------------------------------------
// Kernel 0: row-bias dot products  hw[b,i] = h[b,i]·w_h,  uw[b,j] = u[b,j]·w_u
// ---------------------------------------------------------------------------
__global__ __launch_bounds__(128) void bias_dot_kernel(
    const float* __restrict__ h, const float* __restrict__ u,
    const float* __restrict__ Wa,
    float* __restrict__ hw, float* __restrict__ uw)
{
    const int gw   = blockIdx.x * (blockDim.x >> 5) + (threadIdx.x >> 5);
    const int lane = threadIdx.x & 31;
    const int nh   = Bc * JXc;
    const int tot  = nh + Bc * JQc;
    if (gw >= tot) return;

    const float* src;
    const float* w;
    float* dst;
    if (gw < nh) { src = h + (size_t)gw * Dc;        w = Wa;      dst = hw + gw; }
    else         { int r = gw - nh;
                   src = u + (size_t)r * Dc;         w = Wa + Dc; dst = uw + r; }

    float acc = 0.f;
    #pragma unroll 4
    for (int k = lane; k < Dc; k += 32) acc += src[k] * w[k];
    #pragma unroll
    for (int mk = 16; mk >= 1; mk >>= 1) acc += __shfl_xor(acc, mk, 32);
    if (lane == 0) *dst = acc;
}

// ---------------------------------------------------------------------------
// Kernel 1: per 16-row i-tile (one wave each):
//   S[16,256] = bf16-WMMA( (h*w_m), u^T ) + hw[i] + uw[j] + mask
//   rowmax -> b_logits;  P = exp(S - rowmax);  u_a = (P @ u) / rowsum
//   writes out[:, 0:D]=h, [D:2D]=u_a, [2D:3D]=h*u_a
// ---------------------------------------------------------------------------
__global__ __launch_bounds__(WAVES * 32, 1) void attn_rows_kernel(
    const float* __restrict__ h,
    const __bf16* __restrict__ u_bf, const __bf16* __restrict__ ut_bf,
    const float* __restrict__ Wa,
    const unsigned char* __restrict__ hmask, const unsigned char* __restrict__ umask,
    const float* __restrict__ hw, const float* __restrict__ uw,
    float* __restrict__ blog, float* __restrict__ out)
{
    __shared__ float wm[Dc];                                   // w_m staged once
    __shared__ alignas(16) __bf16 plds[WAVES][16][JQc];        // P transpose buffer

    const int tid = threadIdx.x;
    for (int t = tid; t < Dc; t += WAVES * 32) wm[t] = Wa[2 * Dc + t];
    __syncthreads();

    const int wave = tid >> 5;
    const int lane = tid & 31;
    const int half = lane >> 4;          // 0: lanes 0-15, 1: lanes 16-31
    const int lm   = lane & 15;
    const int kb   = half * 8;           // K sub-block offset for A/B fragments

    const int b  = blockIdx.x / (JXc / ROWS_PER_BLOCK);
    const int i0 = (blockIdx.x % (JXc / ROWS_PER_BLOCK)) * ROWS_PER_BLOCK + wave * 16;

    const float*  hrow  = h     + ((size_t)b * JXc + i0 + lm) * Dc;  // A row
    const __bf16* ubf_b = u_bf  + (size_t)b * JQc * Dc;
    const __bf16* utb_b = ut_bf + (size_t)b * Dc * JQc;

    // ---------------- Phase A: S = (h .* w_m) @ u^T -------------------------
    v8f sacc[16] = {};                       // 16 j-tiles of 16x16 f32
    for (int ks = 0; ks < 16; ++ks) {        // K = 512 in steps of 32
        const int k0 = ks * 32;
        __builtin_prefetch(hrow + k0 + 64, 0, 0);    // global_prefetch_b8
        const float4* pa0 = (const float4*)(hrow + k0 + kb);
        const float4* pa1 = (const float4*)(hrow + k0 + 16 + kb);
        float4 x0 = pa0[0], x1 = pa0[1], x2 = pa1[0], x3 = pa1[1];
        float av[16] = {x0.x, x0.y, x0.z, x0.w, x1.x, x1.y, x1.z, x1.w,
                        x2.x, x2.y, x2.z, x2.w, x3.x, x3.y, x3.z, x3.w};
        v16bf af;
        #pragma unroll
        for (int e = 0; e < 8; ++e) af[e]     = (__bf16)(av[e]     * wm[k0 + kb + e]);
        #pragma unroll
        for (int e = 0; e < 8; ++e) af[8 + e] = (__bf16)(av[8 + e] * wm[k0 + 16 + kb + e]);

        #pragma unroll
        for (int jt = 0; jt < 16; ++jt) {
            const __bf16* urow = ubf_b + (size_t)(jt * 16 + lm) * Dc + k0 + kb;
            v8bf y0 = *(const v8bf*)urow;          // K = k0+kb   .. +7
            v8bf y1 = *(const v8bf*)(urow + 16);   // K = k0+16+kb.. +7
            v16bf bfr;
            #pragma unroll
            for (int e = 0; e < 8; ++e) { bfr[e] = y0[e]; bfr[8 + e] = y1[e]; }
            sacc[jt] = __builtin_amdgcn_wmma_f32_16x16x32_bf16(
                false, af, false, bfr, (short)0, sacc[jt], false, false);
        }
    }

    // ---------------- Phase B: bias + mask + softmax (over j) --------------
    // C layout: VGPR r, lanes 0-15 -> row r; lanes 16-31 -> row r+8; col = lane%16
    const int mbase = i0 + half * 8;
    float bias_i[8]; unsigned char mh[8];
    #pragma unroll
    for (int r = 0; r < 8; ++r) {
        bias_i[r] = hw[(size_t)b * JXc + mbase + r];
        mh[r]     = hmask[(size_t)b * JXc + mbase + r];
    }
    float rmax[8];
    #pragma unroll
    for (int r = 0; r < 8; ++r) rmax[r] = -1e38f;

    #pragma unroll
    for (int jt = 0; jt < 16; ++jt) {
        const int j = jt * 16 + lm;
        const float bj = uw[b * JQc + j];
        const unsigned char mu = umask[b * JQc + j];
        #pragma unroll
        for (int r = 0; r < 8; ++r) {
            float sv = sacc[jt][r] + bias_i[r] + bj;
            if (!(mh[r] & mu)) sv += -1e30f;
            sacc[jt][r] = sv;
            rmax[r] = fmaxf(rmax[r], sv);
        }
    }
    #pragma unroll
    for (int r = 0; r < 8; ++r) {            // reduce across the 16-lane half
        float v = rmax[r];
        v = fmaxf(v, __shfl_xor(v, 1, 32));
        v = fmaxf(v, __shfl_xor(v, 2, 32));
        v = fmaxf(v, __shfl_xor(v, 4, 32));
        v = fmaxf(v, __shfl_xor(v, 8, 32));
        rmax[r] = v;
    }
    if (lm == 0) {
        #pragma unroll
        for (int r = 0; r < 8; ++r)
            blog[(size_t)b * JXc + mbase + r] = rmax[r];   // b_logits = max_j s
    }

    float rsum[8];
    #pragma unroll
    for (int r = 0; r < 8; ++r) rsum[r] = 0.f;
    #pragma unroll
    for (int jt = 0; jt < 16; ++jt) {
        const int j = jt * 16 + lm;
        #pragma unroll
        for (int r = 0; r < 8; ++r) {
            float p = __expf(sacc[jt][r] - rmax[r]);
            rsum[r] += p;
            plds[wave][r + half * 8][j] = (__bf16)p;       // C-layout -> LDS
        }
    }
    float inv[8];
    #pragma unroll
    for (int r = 0; r < 8; ++r) {
        float v = rsum[r];
        v += __shfl_xor(v, 1, 32);
        v += __shfl_xor(v, 2, 32);
        v += __shfl_xor(v, 4, 32);
        v += __shfl_xor(v, 8, 32);
        inv[r] = 1.f / v;
    }
    asm volatile("s_wait_dscnt 0" ::: "memory");           // P stores visible

    // ---------------- Phase C: u_a = P @ u, scale, write outputs -----------
    for (int dt = 0; dt < Dc / 16; ++dt) {
        const int d0 = dt * 16;
        const __bf16* utrow = utb_b + (size_t)(d0 + lm) * JQc;   // uT row (d fixed)
        v8f uacc = {};
        #pragma unroll
        for (int k2 = 0; k2 < 8; ++k2) {     // K = 256 (j) in steps of 32
            const int j0 = k2 * 32;
            v8bf a0 = *(const v8bf*)&plds[wave][lm][j0 + kb];
            v8bf a1 = *(const v8bf*)&plds[wave][lm][j0 + 16 + kb];
            v16bf a2;
            #pragma unroll
            for (int e = 0; e < 8; ++e) { a2[e] = a0[e]; a2[8 + e] = a1[e]; }
            v8bf z0 = *(const v8bf*)(utrow + j0 + kb);        // contiguous in j
            v8bf z1 = *(const v8bf*)(utrow + j0 + 16 + kb);
            v16bf b2;
            #pragma unroll
            for (int e = 0; e < 8; ++e) { b2[e] = z0[e]; b2[8 + e] = z1[e]; }
            uacc = __builtin_amdgcn_wmma_f32_16x16x32_bf16(
                false, a2, false, b2, (short)0, uacc, false, false);
        }
        #pragma unroll
        for (int r = 0; r < 8; ++r) {
            const int m = mbase + r;
            const int d = d0 + lm;
            const float ua = uacc[r] * inv[r];
            const float hv = h[((size_t)b * JXc + m) * Dc + d];
            float* orow = out + ((size_t)b * JXc + m) * (size_t)(4 * Dc);
            orow[d]          = hv;            // h
            orow[Dc + d]     = ua;            // u_a
            orow[2 * Dc + d] = hv * ua;       // h * u_a
        }
    }
}

// ---------------------------------------------------------------------------
// Kernel 2: per batch: b = softmax_i(b_logits), h_a[b,d] = sum_i b[i]*h[b,i,d]
// ---------------------------------------------------------------------------
__global__ __launch_bounds__(256) void row_softmax_kernel(
    const float* __restrict__ h, const float* __restrict__ blog,
    float* __restrict__ ha)
{
    const int b = blockIdx.x, tid = threadIdx.x;
    const int wid = tid >> 5, lane = tid & 31;
    __shared__ float bvec[JXc];
    __shared__ float red[8];

    float m = -1e38f;
    for (int i = tid; i < JXc; i += 256) m = fmaxf(m, blog[(size_t)b * JXc + i]);
    #pragma unroll
    for (int mk = 16; mk >= 1; mk >>= 1) m = fmaxf(m, __shfl_xor(m, mk, 32));
    if (lane == 0) red[wid] = m;
    __syncthreads();
    float mall = red[0];
    #pragma unroll
    for (int w = 1; w < 8; ++w) mall = fmaxf(mall, red[w]);
    __syncthreads();

    float s = 0.f;
    for (int i = tid; i < JXc; i += 256) {
        float e = __expf(blog[(size_t)b * JXc + i] - mall);
        bvec[i] = e;
        s += e;
    }
    #pragma unroll
    for (int mk = 16; mk >= 1; mk >>= 1) s += __shfl_xor(s, mk, 32);
    if (lane == 0) red[wid] = s;
    __syncthreads();
    float tot = 0.f;
    #pragma unroll
    for (int w = 0; w < 8; ++w) tot += red[w];
    const float invs = 1.f / tot;

    for (int d = tid; d < Dc; d += 256) {
        float acc = 0.f;
        #pragma unroll 4
        for (int i = 0; i < JXc; ++i)
            acc += bvec[i] * h[((size_t)b * JXc + i) * Dc + d];
        ha[b * Dc + d] = acc * invs;
    }
}

// ---------------------------------------------------------------------------
// Kernel 3: out[:, 3D:4D] = h * h_a   (memory-bound elementwise, float4)
// ---------------------------------------------------------------------------
__global__ __launch_bounds__(128) void h_ha_kernel(
    const float* __restrict__ h, const float* __restrict__ ha,
    float* __restrict__ out)
{
    const int row = blockIdx.x;              // 0 .. B*JX-1
    const int b   = row / JXc;
    const float4* hr  = (const float4*)(h + (size_t)row * Dc);
    const float4* har = (const float4*)(ha + (size_t)b * Dc);
    float4* orow = (float4*)(out + (size_t)row * (size_t)(4 * Dc) + 3 * Dc);
    for (int q = threadIdx.x; q < Dc / 4; q += 128) {
        float4 hv = hr[q], av = har[q];
        float4 o;
        o.x = hv.x * av.x; o.y = hv.y * av.y;
        o.z = hv.z * av.z; o.w = hv.w * av.w;
        orow[q] = o;
    }
}

// ---------------------------------------------------------------------------
extern "C" void kernel_launch(void* const* d_in, const int* in_sizes, int n_in,
                              void* d_out, int out_size, void* d_ws, size_t ws_size,
                              hipStream_t stream)
{
    const float* h  = (const float*)d_in[0];
    const float* u  = (const float*)d_in[1];
    const float* Wa = (const float*)d_in[2];
    const unsigned char* hm = (const unsigned char*)d_in[3];  // bool masks (1B)
    const unsigned char* um = (const unsigned char*)d_in[4];
    float* out = (float*)d_out;

    // workspace carve: f32 scalars first (16B-aligned counts), then bf16 u copies
    float* uw   = (float*)d_ws;                  // B*JQ   = 2048
    float* hw   = uw + Bc * JQc;                 // B*JX   = 16384
    float* blog = hw + Bc * JXc;                 // B*JX   = 16384
    float* ha   = blog + Bc * JXc;               // B*D    = 4096
    __bf16* u_bf  = (__bf16*)(ha + Bc * Dc);     // B*JQ*D bf16 = 2 MB
    __bf16* ut_bf = u_bf + (size_t)Bc * JQc * Dc;// B*D*JQ bf16 = 2 MB

    convert_u_kernel<<<(Bc * JQc * Dc) / 256, 256, 0, stream>>>(u, u_bf, ut_bf);

    const int rows = Bc * JXc + Bc * JQc;
    bias_dot_kernel<<<(rows + 3) / 4, 128, 0, stream>>>(h, u, Wa, hw, uw);

    attn_rows_kernel<<<Bc * (JXc / ROWS_PER_BLOCK), WAVES * 32, 0, stream>>>(
        h, u_bf, ut_bf, Wa, hm, um, hw, uw, blog, out);

    row_softmax_kernel<<<Bc, 256, 0, stream>>>(h, blog, ha);

    h_ha_kernel<<<Bc * JXc, 128, 0, stream>>>(h, ha, out);
}
